// Transformer_9139690405976
// MI455X (gfx1250) — matrix operand order repaired and checked by reading, not compiled, not run on hardware
//
#include <hip/hip_runtime.h>
#include <hip/hip_bf16.h>
#include <math.h>

#define DEPTH 4
#define DIM   768
#define HEADS 12
#define HD    64
#define CTX   2048
#define BATCH 4
#define ROWS  (BATCH * CTX)   // 8192 token rows

typedef __attribute__((ext_vector_type(8)))  _Float16 h8;
typedef __attribute__((ext_vector_type(16))) _Float16 h16v;
typedef __attribute__((ext_vector_type(8)))  float    f8;

// Async LDS-DMA path (gfx1250 GLOBAL_LOAD_ASYNC_TO_LDS_B128, ASYNCcnt):
// only if this toolchain exposes the builtins; otherwise VGPR staging.
// Probe result: the builtin takes int4* params -> (global int4*, lds int4*, imm, imm).
#if __has_builtin(__builtin_amdgcn_global_load_async_to_lds_b128) && \
    __has_builtin(__builtin_amdgcn_s_wait_asynccnt)
#define ASYNC_LDS 1
typedef int v4i __attribute__((vector_size(16)));
typedef __attribute__((address_space(1))) v4i glb_v4i;
typedef __attribute__((address_space(3))) v4i lds_v4i;
#else
#define ASYNC_LDS 0
#endif

// Concatenate two contiguous 8-half (16B) chunks into one 16-half WMMA fragment.
__device__ __forceinline__ h16v frag_cat(const _Float16* p0, const _Float16* p1) {
  h8 a = *(const h8*)p0;
  h8 b = *(const h8*)p1;
  return __builtin_shufflevector(a, b, 0,1,2,3,4,5,6,7,8,9,10,11,12,13,14,15);
}

__device__ __forceinline__ f8 wmma16(h16v a, h16v b, f8 c) {
  // D = A(16x32 f16) * B(32x16 f16) + C(16x16 f32)
  return __builtin_amdgcn_wmma_f32_16x16x32_f16(false, a, false, b, (short)0, c,
                                                false, false);
}

// ---------------------------------------------------------------------------
// f32 -> f16 weight conversion
// ---------------------------------------------------------------------------
__global__ __launch_bounds__(256) void cvt_kernel(const float* __restrict__ w,
                                                  _Float16* __restrict__ o, int n) {
  int i = blockIdx.x * 256 + threadIdx.x;
  if (i < n) o[i] = (_Float16)w[i];
}

// ---------------------------------------------------------------------------
// x = x_in + pos_emb (broadcast over batch)
// ---------------------------------------------------------------------------
__global__ __launch_bounds__(256) void addpos_kernel(const float* __restrict__ x,
                                                     const float* __restrict__ pos,
                                                     float* __restrict__ xs) {
  size_t i = (size_t)blockIdx.x * 256 + threadIdx.x;
  xs[i] = x[i] + pos[i % ((size_t)CTX * DIM)];
}

// ---------------------------------------------------------------------------
// LayerNorm over DIM=768, one block (8 waves) per row, f16 output
// ---------------------------------------------------------------------------
__global__ __launch_bounds__(256) void ln_kernel(const float* __restrict__ x,
                                                 const float* __restrict__ g,
                                                 const float* __restrict__ bta,
                                                 _Float16* __restrict__ out) {
  int row = blockIdx.x;
  int t = threadIdx.x;
  const float* xr = x + (size_t)row * DIM;
  float v0 = xr[t], v1 = xr[t + 256], v2 = xr[t + 512];
  float s = v0 + v1 + v2;
  float s2 = v0 * v0 + v1 * v1 + v2 * v2;
  for (int off = 16; off; off >>= 1) {
    s  += __shfl_xor(s,  off, 32);
    s2 += __shfl_xor(s2, off, 32);
  }
  __shared__ float ps[8], ps2[8];
  int w = t >> 5, lane = t & 31;
  if (lane == 0) { ps[w] = s; ps2[w] = s2; }
  __syncthreads();
  float ts = 0.f, ts2 = 0.f;
  for (int i = 0; i < 8; ++i) { ts += ps[i]; ts2 += ps2[i]; }
  float mean = ts * (1.f / DIM);
  float var  = ts2 * (1.f / DIM) - mean * mean;
  float rstd = rsqrtf(var + 1e-5f);
  _Float16* orow = out + (size_t)row * DIM;
  orow[t]       = (_Float16)((v0 - mean) * rstd * g[t]       + bta[t]);
  orow[t + 256] = (_Float16)((v1 - mean) * rstd * g[t + 256] + bta[t + 256]);
  orow[t + 512] = (_Float16)((v2 - mean) * rstd * g[t + 512] + bta[t + 512]);
}

// ---------------------------------------------------------------------------
// Generic WMMA GEMM: C[M,N] = act(A[M,K] @ B[K,N] + bias) (+ residual)
// A,B f16 row-major. Block = 256 threads (8 waves, 4x2 wave grid), tile
// BM=128, BN=128, BK=32. Each wave owns a 32x64 output tile = 2x4 WMMA tiles
// (8 v_wmma per K-step). A-tile: double-buffered async LDS-DMA
// (global_load_async_to_lds_b128, ASYNCcnt) overlapping a full K-step of
// WMMA. B-tile staged transposed via VGPRs so every fragment fetch is a
// pair of contiguous ds_load_b128.
// ---------------------------------------------------------------------------
template <bool HAS_BIAS, bool DO_GELU, bool HAS_RESID>
__global__ __launch_bounds__(256) void gemm_kernel(
    const _Float16* __restrict__ A, const _Float16* __restrict__ Bw,
    const float* __restrict__ bias, const float* resid,
    float* out32, _Float16* out16, int M, int N, int K) {
#if ASYNC_LDS
  __shared__ __align__(32) _Float16 As[2][128 * 32];  // [buf][m][k]
#else
  __shared__ __align__(32) _Float16 As[1][128 * 32];  // [m][k]
#endif
  __shared__ __align__(32) _Float16 Bs[128 * 32];     // [n][k] (transposed)
  int m0 = blockIdx.x * 128;
  int n0 = blockIdx.y * 128;
  int t = threadIdx.x;
  int lane = t & 31, w = t >> 5;
  int wm = w >> 1, wn = w & 1;          // 4 (M) x 2 (N) wave grid
  int hi = lane >> 4, ln16 = lane & 15;

  // Per-thread staging coordinates (two 8-half chunks per operand).
  int linA0 = t * 8,            linA1 = (t + 256) * 8;
  int rowA0 = linA0 >> 5,       colA0 = linA0 & 31;
  int rowA1 = linA1 >> 5,       colA1 = linA1 & 31;
  int krB0  = linA0 >> 7,       colB0 = linA0 & 127;   // B tile 32(k) x 128(n)
  int krB1  = linA1 >> 7,       colB1 = linA1 & 127;

  const f8 zf = {0.f, 0.f, 0.f, 0.f, 0.f, 0.f, 0.f, 0.f};
  f8 acc[2][4] = {{zf, zf, zf, zf}, {zf, zf, zf, zf}};

  h8 rb0, rb1;
  auto stage_B = [&](int kt) {
    rb0 = *(const h8*)(Bw + (size_t)(kt + krB0) * N + n0 + colB0);
    rb1 = *(const h8*)(Bw + (size_t)(kt + krB1) * N + n0 + colB1);
  };
#if ASYNC_LDS
  auto async_A = [&](int kt, int buf) {
    __builtin_amdgcn_global_load_async_to_lds_b128(
        (glb_v4i*)(A + (size_t)(m0 + rowA0) * K + kt + colA0),
        (lds_v4i*)&As[buf][rowA0 * 32 + colA0], 0, 0);
    __builtin_amdgcn_global_load_async_to_lds_b128(
        (glb_v4i*)(A + (size_t)(m0 + rowA1) * K + kt + colA1),
        (lds_v4i*)&As[buf][rowA1 * 32 + colA1], 0, 0);
  };
  async_A(0, 0);
#else
  h8 ra0, ra1;
  auto stage_A = [&](int kt) {
    ra0 = *(const h8*)(A + (size_t)(m0 + rowA0) * K + kt + colA0);
    ra1 = *(const h8*)(A + (size_t)(m0 + rowA1) * K + kt + colA1);
  };
  stage_A(0);
#endif
  stage_B(0);

  int buf = 0;
  for (int kt = 0; kt < K; kt += 32, buf ^= 1) {
    __syncthreads();   // all waves done reading LDS of previous step
    for (int i = 0; i < 8; ++i) Bs[(colB0 + i) * 32 + krB0] = rb0[i];
    for (int i = 0; i < 8; ++i) Bs[(colB1 + i) * 32 + krB1] = rb1[i];
#if ASYNC_LDS
    __builtin_amdgcn_s_wait_asynccnt(0);         // As[buf] DMA complete
    if (kt + 32 < K) {
      async_A(kt + 32, buf ^ 1);                 // DMA next tile during WMMA
      stage_B(kt + 32);
    }
#else
    *(h8*)&As[0][rowA0 * 32 + colA0] = ra0;
    *(h8*)&As[0][rowA1 * 32 + colA1] = ra1;
    if (kt + 32 < K) { stage_A(kt + 32); stage_B(kt + 32); }
#endif
    __syncthreads();   // publish As/Bs to all waves

#if ASYNC_LDS
    const _Float16* Ab = &As[buf][0];
#else
    const _Float16* Ab = &As[0][0];
#endif
    h16v af[2], bf[4];
    for (int mt = 0; mt < 2; ++mt) {
      // A frag: halves = K 16*hi .. 16*hi+15
      const _Float16* ap = &Ab[(wm * 32 + mt * 16 + ln16) * 32 + 16 * hi];
      af[mt] = frag_cat(ap, ap + 8);
    }
    for (int nt = 0; nt < 4; ++nt) {
      // B frag: halves = K 16*hi .. 16*hi+15, contiguous in transposed layout
      const _Float16* bp = &Bs[(wn * 64 + nt * 16 + ln16) * 32 + 16 * hi];
      bf[nt] = frag_cat(bp, bp + 8);
    }
    for (int mt = 0; mt < 2; ++mt)
      for (int nt = 0; nt < 4; ++nt)
        acc[mt][nt] = wmma16(af[mt], bf[nt], acc[mt][nt]);
  }

  // epilogue: C layout lane (r,hi,ln16) -> row = r+8*hi, col = ln16
  for (int mt = 0; mt < 2; ++mt)
    for (int nt = 0; nt < 4; ++nt) {
      int col = n0 + wn * 64 + nt * 16 + ln16;
      float bval = 0.f;
      if constexpr (HAS_BIAS) bval = bias[col];
      for (int r = 0; r < 8; ++r) {
        int row = m0 + wm * 32 + mt * 16 + r + 8 * hi;
        float v2 = acc[mt][nt][r] + bval;
        if constexpr (DO_GELU) v2 = 0.5f * v2 * (1.f + erff(v2 * 0.70710678118654752f));
        if constexpr (HAS_RESID) v2 += resid[(size_t)row * N + col];
        if (out32) out32[(size_t)row * N + col] = v2;
        if (out16) out16[(size_t)row * N + col] = (_Float16)v2;
      }
    }
}

// ---------------------------------------------------------------------------
// Flash attention: one block (8 waves, 256 threads) per (b, head, 128 q-rows).
// Each wave owns 16 q rows; a staged 32-key K/V tile feeds all 8 waves.
// S = Q K^T via WMMA, online softmax in C-layout registers, P staged through
// per-wave LDS to build the A-fragment, O += P V via WMMA vs transposed V.
// ---------------------------------------------------------------------------
__global__ __launch_bounds__(256) void attn_kernel(const _Float16* __restrict__ q,
                                                   const _Float16* __restrict__ k,
                                                   const _Float16* __restrict__ v,
                                                   _Float16* __restrict__ o) {
  __shared__ __align__(32) _Float16 Kt[32 * 64];     // [key][d]
  __shared__ __align__(32) _Float16 Vt[64 * 32];     // [d][key] (transposed)
  __shared__ __align__(32) _Float16 Pb[8][16][32];   // per-wave P tile [qrow][key]

  int blk = blockIdx.x;
  int b   = blk / (HEADS * (CTX / 128));
  int rem = blk % (HEADS * (CTX / 128));
  int hd  = rem / (CTX / 128);
  int qb  = rem % (CTX / 128);

  int t = threadIdx.x;
  int w = t >> 5, lane = t & 31, hi = lane >> 4, ln16 = lane & 15;
  size_t baseRow = (size_t)b * CTX;

  // Q fragments (held in registers for whole K loop): 16 rows x 64 d = 2 frags
  int qrow = qb * 128 + w * 16 + ln16;
  const _Float16* qp = q + (baseRow + qrow) * DIM + hd * HD;
  h16v qa0 = frag_cat(qp + 16 * hi,       qp + 8 + 16 * hi);
  h16v qa1 = frag_cat(qp + 32 + 16 * hi,  qp + 40 + 16 * hi);

  const f8 zf = {0.f, 0.f, 0.f, 0.f, 0.f, 0.f, 0.f, 0.f};
  f8 oacc[4] = {zf, zf, zf, zf};
  float mrow[8], lrow[8];
  for (int r = 0; r < 8; ++r) { mrow[r] = -1e30f; lrow[r] = 0.f; }

  for (int kt = 0; kt < CTX / 32; ++kt) {
    __syncthreads();
    // stage K tile (row-major) and V tile (transposed); one 8-half chunk each
    {
      int lin = t * 8;
      int key = lin >> 6;    // 0..31
      int col = lin & 63;    // 0..63 step 8
      const _Float16* ks = k + (baseRow + kt * 32 + key) * DIM + hd * HD + col;
      *(h8*)&Kt[key * 64 + col] = *(const h8*)ks;
      const _Float16* vs = v + (baseRow + kt * 32 + key) * DIM + hd * HD + col;
      h8 vv = *(const h8*)vs;
      for (int i = 0; i < 8; ++i) Vt[(col + i) * 32 + key] = vv[i];
    }
    __syncthreads();

    // S = Q K^T  (two 16x16 tiles covering 32 keys), scaled by 1/sqrt(64)
    f8 s0 = zf, s1 = zf;
    {
      const _Float16* kb0 = &Kt[(0 + ln16) * 64 + 16 * hi];
      s0 = wmma16(qa0, frag_cat(kb0, kb0 + 8), s0);
      s0 = wmma16(qa1, frag_cat(kb0 + 32, kb0 + 40), s0);
      const _Float16* kb1 = &Kt[(16 + ln16) * 64 + 16 * hi];
      s1 = wmma16(qa0, frag_cat(kb1, kb1 + 8), s1);
      s1 = wmma16(qa1, frag_cat(kb1 + 32, kb1 + 40), s1);
    }

    // online softmax per q row (row = r + 8*hi; 16 lanes span the 32 keys)
    const float scale = 0.125f;
    for (int r = 0; r < 8; ++r) {
      float a0 = s0[r] * scale, a1 = s1[r] * scale;
      float mx = fmaxf(a0, a1);
      mx = fmaxf(mx, __shfl_xor(mx, 1, 32));
      mx = fmaxf(mx, __shfl_xor(mx, 2, 32));
      mx = fmaxf(mx, __shfl_xor(mx, 4, 32));
      mx = fmaxf(mx, __shfl_xor(mx, 8, 32));
      float mnew = fmaxf(mrow[r], mx);
      float p0 = __expf(a0 - mnew), p1 = __expf(a1 - mnew);
      float psum = p0 + p1;
      psum += __shfl_xor(psum, 1, 32);
      psum += __shfl_xor(psum, 2, 32);
      psum += __shfl_xor(psum, 4, 32);
      psum += __shfl_xor(psum, 8, 32);
      float corr = __expf(mrow[r] - mnew);
      lrow[r] = lrow[r] * corr + psum;
      mrow[r] = mnew;
      for (int dt = 0; dt < 4; ++dt) oacc[dt][r] *= corr;
      Pb[w][r + 8 * hi][ln16]      = (_Float16)p0;
      Pb[w][r + 8 * hi][ln16 + 16] = (_Float16)p1;
    }

    // O += P V  (P 16x32, V 32x64 in 4 n-tiles). Same-wave LDS RAW is
    // in-order on CDNA5, no block barrier needed (Pb region is per-wave).
    const _Float16* pp = &Pb[w][ln16][0];
    h16v pf = frag_cat(pp + 16 * hi, pp + 8 + 16 * hi);
    for (int dt = 0; dt < 4; ++dt) {
      const _Float16* vb = &Vt[(dt * 16 + ln16) * 32 + 16 * hi];
      oacc[dt] = wmma16(pf, frag_cat(vb, vb + 8), oacc[dt]);
    }
  }

  // normalize by row sum and store f16
  for (int r = 0; r < 8; ++r) {
    float inv = 1.f / lrow[r];
    int orow = qb * 128 + w * 16 + r + 8 * hi;
    _Float16* op = o + (baseRow + orow) * DIM + hd * HD;
    for (int dt = 0; dt < 4; ++dt)
      op[dt * 16 + ln16] = (_Float16)(oacc[dt][r] * inv);
  }
}

// ---------------------------------------------------------------------------
// Host orchestration
// ---------------------------------------------------------------------------
extern "C" void kernel_launch(void* const* d_in, const int* in_sizes, int n_in,
                              void* d_out, int out_size, void* d_ws, size_t ws_size,
                              hipStream_t stream) {
  (void)in_sizes; (void)n_in; (void)out_size; (void)ws_size;
  const float* x_in = (const float*)d_in[0];
  const float* pos  = (const float*)d_in[1];
  const float* ln1g = (const float*)d_in[2];
  const float* ln1b = (const float*)d_in[3];
  const float* Wq   = (const float*)d_in[4];
  const float* Wk   = (const float*)d_in[5];
  const float* Wv   = (const float*)d_in[6];
  const float* Wo   = (const float*)d_in[7];
  const float* ln2g = (const float*)d_in[8];
  const float* ln2b = (const float*)d_in[9];
  const float* W1   = (const float*)d_in[10];
  const float* b1   = (const float*)d_in[11];
  const float* W2   = (const float*)d_in[12];
  const float* b2   = (const float*)d_in[13];

  const size_t actF32 = (size_t)ROWS * DIM * sizeof(float);
  const size_t actF16 = (size_t)ROWS * DIM * sizeof(_Float16);
  const size_t wF16   = (size_t)DEPTH * DIM * DIM * sizeof(_Float16);

  char* p = (char*)d_ws;
  float*    xs   = (float*)p;    p += actF32;
  _Float16* h16b = (_Float16*)p; p += actF16;
  _Float16* q16  = (_Float16*)p; p += actF16;
  _Float16* k16  = (_Float16*)p; p += actF16;
  _Float16* v16  = (_Float16*)p; p += actF16;
  _Float16* o16  = (_Float16*)p; p += actF16;
  _Float16* g16  = (_Float16*)p; p += actF16;
  _Float16* wq16 = (_Float16*)p; p += wF16;
  _Float16* wk16 = (_Float16*)p; p += wF16;
  _Float16* wv16 = (_Float16*)p; p += wF16;
  _Float16* wo16 = (_Float16*)p; p += wF16;
  _Float16* w116 = (_Float16*)p; p += wF16;
  _Float16* w216 = (_Float16*)p; p += wF16;

  const int welems = DEPTH * DIM * DIM;
  const int wgrid = welems / 256;
  cvt_kernel<<<wgrid, 256, 0, stream>>>(Wq, wq16, welems);
  cvt_kernel<<<wgrid, 256, 0, stream>>>(Wk, wk16, welems);
  cvt_kernel<<<wgrid, 256, 0, stream>>>(Wv, wv16, welems);
  cvt_kernel<<<wgrid, 256, 0, stream>>>(Wo, wo16, welems);
  cvt_kernel<<<wgrid, 256, 0, stream>>>(W1, w116, welems);
  cvt_kernel<<<wgrid, 256, 0, stream>>>(W2, w216, welems);

  addpos_kernel<<<(ROWS * DIM) / 256, 256, 0, stream>>>(x_in, pos, xs);

  dim3 ggrid(ROWS / 128, DIM / 128);   // 64 x 6
  for (int l = 0; l < DEPTH; ++l) {
    size_t woff = (size_t)l * DIM * DIM;
    // attention block
    ln_kernel<<<ROWS, 256, 0, stream>>>(xs, ln1g + l * DIM, ln1b + l * DIM, h16b);
    gemm_kernel<false, false, false><<<ggrid, 256, 0, stream>>>(
        h16b, wq16 + woff, nullptr, nullptr, nullptr, q16, ROWS, DIM, DIM);
    gemm_kernel<false, false, false><<<ggrid, 256, 0, stream>>>(
        h16b, wk16 + woff, nullptr, nullptr, nullptr, k16, ROWS, DIM, DIM);
    gemm_kernel<false, false, false><<<ggrid, 256, 0, stream>>>(
        h16b, wv16 + woff, nullptr, nullptr, nullptr, v16, ROWS, DIM, DIM);
    attn_kernel<<<BATCH * HEADS * (CTX / 128), 256, 0, stream>>>(q16, k16, v16, o16);
    gemm_kernel<false, false, true><<<ggrid, 256, 0, stream>>>(
        o16, wo16 + woff, nullptr, xs, xs, nullptr, ROWS, DIM, DIM);
    // feed-forward block
    ln_kernel<<<ROWS, 256, 0, stream>>>(xs, ln2g + l * DIM, ln2b + l * DIM, h16b);
    gemm_kernel<true, true, false><<<ggrid, 256, 0, stream>>>(
        h16b, w116 + woff, b1 + l * DIM, nullptr, nullptr, g16, ROWS, DIM, DIM);
    float* outp = (l == DEPTH - 1) ? (float*)d_out : xs;
    gemm_kernel<true, false, true><<<ggrid, 256, 0, stream>>>(
        g16, w216 + woff, b2 + l * DIM, xs, outp, nullptr, ROWS, DIM, DIM);
  }
}
